// SparseAttnLayer_89807766159963
// MI455X (gfx1250) — compile-verified, matrix operand order checked by reading, not run on hardware
//
#include <hip/hip_runtime.h>
#include <cstddef>
#include <cstdint>

typedef __bf16 bf16;
typedef __attribute__((ext_vector_type(16))) __bf16 v16bf;
typedef __attribute__((ext_vector_type(8)))  __bf16 v8bf;
typedef __attribute__((ext_vector_type(8)))  float  v8f;

#define EPI_BF16 0
#define EPI_GELU 1
#define EPI_F32  2

static const long long ELEMS = 67108864ll;   // 8 * 512 * 128 * 128

// ---------------------------------------------------------------------------
// WMMA fragment load: lane l -> row/col (l & 15), K-chunk ((l >> 4) * 16),
// 16 contiguous bf16 from LDS (two 16B-aligned v8bf loads).
// ---------------------------------------------------------------------------
__device__ __forceinline__ v16bf ld_frag(const bf16* p0, int stride) {
  const int l = threadIdx.x & 31;
  const bf16* p = p0 + (l & 15) * stride + ((l >> 4) << 4);
  v8bf lo = *reinterpret_cast<const v8bf*>(p);
  v8bf hi = *reinterpret_cast<const v8bf*>(p + 8);
  v16bf r;
#pragma unroll
  for (int i = 0; i < 8; ++i) { r[i] = lo[i]; r[i + 8] = hi[i]; }
  return r;
}

// ---------------------------------------------------------------------------
// Generic bf16 GEMM: C[M,N] = A[M,K] @ BT[N,K]^T + bias, fused epilogue.
// 256 threads = 8 waves; tile 128x128; wave computes 32x64 (2x4 WMMA frags).
// A/B tiles staged via GLOBAL_LOAD_ASYNC_TO_LDS_B128 (ASYNCcnt path):
// each thread owns a 32B slice of each tile (2 x b128 per matrix).
// ---------------------------------------------------------------------------
__global__ __launch_bounds__(256)
void swin_gemm_bf16(const bf16* __restrict__ A, const bf16* __restrict__ BT,
                    const float* __restrict__ bias, void* __restrict__ Cout,
                    int M, int N, int K, int epi)
{
  __shared__ bf16 As[128][40];
  __shared__ bf16 Bs[128][40];

  const int tid  = threadIdx.x;
  const int lane = tid & 31;
  const int wave = tid >> 5;
  const int wm = wave & 3;        // 4 wave-rows  * 32
  const int wn = wave >> 2;       // 2 wave-cols  * 64
  const int rowBase = blockIdx.y * 128;
  const int colBase = blockIdx.x * 128;

  const int ldRow = tid >> 1;          // 0..127
  const int ldK   = (tid & 1) << 4;    // 0 or 16

  const bf16* aPtr = A  + (size_t)(rowBase + ldRow) * K + ldK;
  const bf16* bPtr = BT + (size_t)(colBase + ldRow) * K + ldK;

  // LDS byte addresses (addr[31:0] of the flat pointer = addrspace(3) offset)
  const uint32_t ldsA = (uint32_t)(uintptr_t)&As[ldRow][ldK];
  const uint32_t ldsB = (uint32_t)(uintptr_t)&Bs[ldRow][ldK];

  v8f acc[2][4] = {};

  for (int k0 = 0; k0 < K; k0 += 32) {
    const uint64_t ga = (uint64_t)(uintptr_t)(aPtr + k0);
    const uint64_t gb = (uint64_t)(uintptr_t)(bPtr + k0);
    if (k0 + 32 < K) {                       // hint next K tile (global_prefetch_b8)
      __builtin_prefetch(aPtr + k0 + 32, 0, 1);
      __builtin_prefetch(bPtr + k0 + 32, 0, 1);
    }
    __syncthreads();   // all readers of the previous tile are done
    // async cache->LDS staging; INST_OFFSET applies to both LDS and global addr
    asm volatile(
        "global_load_async_to_lds_b128 %0, %2, off\n\t"
        "global_load_async_to_lds_b128 %0, %2, off offset:16\n\t"
        "global_load_async_to_lds_b128 %1, %3, off\n\t"
        "global_load_async_to_lds_b128 %1, %3, off offset:16"
        :: "v"(ldsA), "v"(ldsB), "v"(ga), "v"(gb) : "memory");
    asm volatile("s_wait_asynccnt 0" ::: "memory");
    __syncthreads();   // tile visible to all waves

    v16bf af[2], bfr[4];
#pragma unroll
    for (int mi = 0; mi < 2; ++mi) af[mi]  = ld_frag(&As[wm * 32 + mi * 16][0], 40);
#pragma unroll
    for (int ni = 0; ni < 4; ++ni) bfr[ni] = ld_frag(&Bs[wn * 64 + ni * 16][0], 40);
#pragma unroll
    for (int mi = 0; mi < 2; ++mi)
#pragma unroll
      for (int ni = 0; ni < 4; ++ni)
        acc[mi][ni] = __builtin_amdgcn_wmma_f32_16x16x32_bf16(
            false, af[mi], false, bfr[ni], (short)0, acc[mi][ni], false, false);
  }

  // epilogue: element (r, lane) -> M = r + 8*(lane>>4), N = lane&15
#pragma unroll
  for (int mi = 0; mi < 2; ++mi)
#pragma unroll
    for (int ni = 0; ni < 4; ++ni) {
      const int col = colBase + wn * 64 + ni * 16 + (lane & 15);
      const float bcol = bias[col];
#pragma unroll
      for (int r = 0; r < 8; ++r) {
        const int row = rowBase + wm * 32 + mi * 16 + r + ((lane >> 4) << 3);
        float v = acc[mi][ni][r] + bcol;
        if (epi == EPI_GELU) v = 0.5f * v * (1.0f + erff(v * 0.70710678118f));
        if (epi == EPI_F32)
          reinterpret_cast<float*>(Cout)[(size_t)row * N + col] = v;
        else
          reinterpret_cast<bf16*>(Cout)[(size_t)row * N + col] = (bf16)v;
      }
    }
}

// ---------------------------------------------------------------------------
// Window attention: one wave per (window, head). N=64 tokens, HD=32.
// QK^T: 4x4 WMMA tiles (K=32); softmax w/ relative-position bias; PV: 4x2, K=64.
// ---------------------------------------------------------------------------
__global__ __launch_bounds__(32)
void swin_attn(const bf16* __restrict__ qkv, const float* __restrict__ relT,
               bf16* __restrict__ outTok)
{
  __shared__ bf16  Qs[64][40];
  __shared__ bf16  Ks[64][40];
  __shared__ bf16  VTs[32][72];   // V transposed: [d][m]
  __shared__ float Ss[64][64];
  __shared__ bf16  Ps[64][72];

  const int win  = blockIdx.x;    // 0..2047
  const int head = blockIdx.y;    // 0..15
  const int lane = threadIdx.x;
  const size_t tokBase = (size_t)win * 64;
  const int qcol = head * 32;

  for (int n = 0; n < 64; ++n) {   // coalesced 64B per row per matrix
    const size_t base = (tokBase + n) * 1536 + qcol;
    Qs[n][lane]  = qkv[base + lane];
    Ks[n][lane]  = qkv[base + 512 + lane];
    VTs[lane][n] = qkv[base + 1024 + lane];
  }
  __syncthreads();

  v16bf qf[4], kf[4];
#pragma unroll
  for (int i = 0; i < 4; ++i) {
    qf[i] = ld_frag(&Qs[i * 16][0], 40);
    kf[i] = ld_frag(&Ks[i * 16][0], 40);
  }
  v8f s[4][4] = {};
#pragma unroll
  for (int ni = 0; ni < 4; ++ni)
#pragma unroll
    for (int mi = 0; mi < 4; ++mi)
      s[ni][mi] = __builtin_amdgcn_wmma_f32_16x16x32_bf16(
          false, qf[ni], false, kf[mi], (short)0, s[ni][mi], false, false);

#pragma unroll
  for (int ni = 0; ni < 4; ++ni)
#pragma unroll
    for (int mi = 0; mi < 4; ++mi)
#pragma unroll
      for (int r = 0; r < 8; ++r) {
        const int n = ni * 16 + r + ((lane >> 4) << 3);
        const int m = mi * 16 + (lane & 15);
        Ss[n][m] = s[ni][mi][r] * 0.17677669529663689f;   // 1/sqrt(32)
      }
  __syncthreads();

  // softmax over m with Swin relative-position bias; lane handles rows lane, lane+32
  for (int rr = 0; rr < 2; ++rr) {
    const int n = lane + rr * 32;
    const int i1 = n >> 3, j1 = n & 7;
    float mx = -3.0e38f;
    for (int m = 0; m < 64; ++m) {
      const int i2 = m >> 3, j2 = m & 7;
      const int ridx = (i1 - i2 + 7) * 15 + (j1 - j2 + 7);
      const float v = Ss[n][m] + relT[ridx * 16 + head];
      Ss[n][m] = v;
      mx = fmaxf(mx, v);
    }
    float sum = 0.f;
    for (int m = 0; m < 64; ++m) sum += __expf(Ss[n][m] - mx);
    const float inv = 1.0f / sum;
    for (int m = 0; m < 64; ++m) Ps[n][m] = (bf16)(__expf(Ss[n][m] - mx) * inv);
  }
  __syncthreads();

  v8f o[4][2] = {};
#pragma unroll
  for (int kc = 0; kc < 2; ++kc) {
    v16bf pf[4], vf[2];
#pragma unroll
    for (int ni = 0; ni < 4; ++ni) pf[ni] = ld_frag(&Ps[ni * 16][kc * 32], 72);
#pragma unroll
    for (int di = 0; di < 2; ++di) vf[di] = ld_frag(&VTs[di * 16][kc * 32], 72);
#pragma unroll
    for (int ni = 0; ni < 4; ++ni)
#pragma unroll
      for (int di = 0; di < 2; ++di)
        o[ni][di] = __builtin_amdgcn_wmma_f32_16x16x32_bf16(
            false, pf[ni], false, vf[di], (short)0, o[ni][di], false, false);
  }

#pragma unroll
  for (int ni = 0; ni < 4; ++ni)
#pragma unroll
    for (int di = 0; di < 2; ++di)
#pragma unroll
      for (int r = 0; r < 8; ++r) {
        const int n = ni * 16 + r + ((lane >> 4) << 3);
        const int d = di * 16 + (lane & 15);
        outTok[(tokBase + n) * 512 + qcol + d] = (bf16)o[ni][di][r];
      }
}

// ---------------------------------------------------------------------------
// Elementwise / layout kernels
// ---------------------------------------------------------------------------
__global__ void swin_wtrans(const float* __restrict__ W, bf16* __restrict__ WT,
                            int K, int N) {                 // W[K,N] -> WT[N,K] bf16
  long long idx = (long long)blockIdx.x * blockDim.x + threadIdx.x;
  if (idx >= (long long)K * N) return;
  const int n = (int)(idx / K), k = (int)(idx % K);
  WT[idx] = (bf16)W[(size_t)k * N + n];
}

__global__ void swin_wconv(const float* __restrict__ W, bf16* __restrict__ WT,
                           long long total) {               // already [N,K] (conv 'oc')
  long long idx = (long long)blockIdx.x * blockDim.x + threadIdx.x;
  if (idx >= total) return;
  WT[idx] = (bf16)W[idx];
}

__global__ void swin_bn_part(const float* __restrict__ x, const float* __restrict__ g,
                             const float* __restrict__ b, const float* __restrict__ mu,
                             const float* __restrict__ var, bf16* __restrict__ xw) {
  long long idx = (long long)blockIdx.x * blockDim.x + threadIdx.x;
  if (idx >= ELEMS) return;
  const int c = (int)(idx & 511);
  const long long tok = idx >> 9;
  const int n = (int)(tok & 63);
  const long long win = tok >> 6;
  const int bb = (int)(win >> 8);
  const int wg = (int)(win & 255);
  const int h = (wg >> 4) * 8 + (n >> 3);
  const int w = (wg & 15) * 8 + (n & 7);
  const float v = x[(((size_t)bb * 512 + c) * 128 + h) * 128 + w];
  const float inv = rsqrtf(var[c] + 1e-5f);
  xw[idx] = (bf16)((v - mu[c]) * inv * g[c] + b[c]);
}

__global__ void swin_resid(const float* __restrict__ x, const float* __restrict__ proj,
                           float* __restrict__ out) {   // torch .view window_reverse
  long long idx = (long long)blockIdx.x * blockDim.x + threadIdx.x;
  if (idx >= ELEMS) return;
  const int w  = (int)(idx & 127);
  const int h  = (int)((idx >> 7) & 127);
  const int c  = (int)((idx >> 14) & 511);
  const int bb = (int)(idx >> 23);
  const size_t src = ((((((size_t)bb * 512 + c) * 16 + (h >> 3)) * 16 + (w >> 3)) * 8
                        + (h & 7)) * 8 + (w & 7));
  out[idx] = x[idx] + proj[src];
}

__global__ void swin_bn_tok(const float* __restrict__ x1, const float* __restrict__ g,
                            const float* __restrict__ b, const float* __restrict__ mu,
                            const float* __restrict__ var, bf16* __restrict__ xt) {
  long long idx = (long long)blockIdx.x * blockDim.x + threadIdx.x;
  if (idx >= ELEMS) return;
  const int c = (int)(idx & 511);
  const long long t = idx >> 9;
  const int w  = (int)(t & 127);
  const int h  = (int)((t >> 7) & 127);
  const int bb = (int)(t >> 14);
  const float v = x1[(((size_t)bb * 512 + c) * 128 + h) * 128 + w];
  const float inv = rsqrtf(var[c] + 1e-5f);
  xt[idx] = (bf16)((v - mu[c]) * inv * g[c] + b[c]);
}

__global__ void swin_final_add(const bf16* __restrict__ h2, float* __restrict__ out) {
  long long idx = (long long)blockIdx.x * blockDim.x + threadIdx.x;
  if (idx >= ELEMS) return;
  const int w  = (int)(idx & 127);
  const int h  = (int)((idx >> 7) & 127);
  const int c  = (int)((idx >> 14) & 511);
  const int bb = (int)(idx >> 23);
  const size_t t = (size_t)bb * 16384 + (size_t)h * 128 + w;
  out[idx] += (float)h2[t * 512 + c];
}

// ---------------------------------------------------------------------------
extern "C" void kernel_launch(void* const* d_in, const int* in_sizes, int n_in,
                              void* d_out, int out_size, void* d_ws, size_t ws_size,
                              hipStream_t stream) {
  (void)in_sizes; (void)n_in; (void)out_size; (void)ws_size;
  const float* x      = (const float*)d_in[0];
  const float* qkv_w  = (const float*)d_in[1];
  const float* qkv_b  = (const float*)d_in[2];
  const float* proj_w = (const float*)d_in[3];
  const float* proj_b = (const float*)d_in[4];
  const float* relT   = (const float*)d_in[5];
  const float* bn1_g  = (const float*)d_in[6];
  const float* bn1_b  = (const float*)d_in[7];
  const float* bn1_m  = (const float*)d_in[8];
  const float* bn1_v  = (const float*)d_in[9];
  const float* w1     = (const float*)d_in[10];
  const float* b1     = (const float*)d_in[11];
  const float* w2     = (const float*)d_in[12];
  const float* b2     = (const float*)d_in[13];
  const float* bn2_g  = (const float*)d_in[14];
  const float* bn2_b  = (const float*)d_in[15];
  const float* bn2_m  = (const float*)d_in[16];
  const float* bn2_v  = (const float*)d_in[17];
  float* out = (float*)d_out;

  char* ws = (char*)d_ws;
  bf16* WqkvT  = (bf16*)(ws);                                 // 1.5 MB
  bf16* WprojT = (bf16*)(ws + (2ull << 20));                  // 0.5 MB
  bf16* W1c    = (bf16*)(ws + (3ull << 20));                  // 0.5 MB
  bf16* W2c    = (bf16*)(ws + (4ull << 20));                  // 0.5 MB
  bf16* bufA   = (bf16*)(ws + (8ull << 20));                  // 128 MB: xw, later xn2 tokens
  char* bufB   =          ws + (8ull << 20) + (132ull << 20); // 384 MB: qkv -> projF32 -> h1
  bf16* bufC   = (bf16*)(ws + (8ull << 20) + (132ull << 20) + (388ull << 20)); // 128 MB

  bf16*  qkvBuf = (bf16*)bufB;
  float* projF  = (float*)bufB;
  bf16*  h1     = (bf16*)bufB;

  const int TB = 256;
  const int eb = (int)(ELEMS / TB);   // 262144 blocks, exact

  // weights -> bf16 "BT" layout (N x K row-major)
  swin_wtrans<<<(1536 * 512 + 255) / 256, TB, 0, stream>>>(qkv_w, WqkvT, 512, 1536);
  swin_wtrans<<<(512 * 512 + 255) / 256, TB, 0, stream>>>(proj_w, WprojT, 512, 512);
  swin_wconv<<<(512 * 512 + 255) / 256, TB, 0, stream>>>(w1, W1c, 512ll * 512);
  swin_wconv<<<(512 * 512 + 255) / 256, TB, 0, stream>>>(w2, W2c, 512ll * 512);

  // BN1 + window partition -> token-major bf16
  swin_bn_part<<<eb, TB, 0, stream>>>(x, bn1_g, bn1_b, bn1_m, bn1_v, bufA);
  // QKV GEMM: 131072 x 1536 = (131072 x 512) @ (512 x 1536)
  swin_gemm_bf16<<<dim3(12, 1024), 256, 0, stream>>>(bufA, WqkvT, qkv_b, qkvBuf,
                                                     131072, 1536, 512, EPI_BF16);
  // window attention per (window, head)
  swin_attn<<<dim3(2048, 16), 32, 0, stream>>>(qkvBuf, relT, bufC);
  // proj GEMM -> fp32
  swin_gemm_bf16<<<dim3(4, 1024), 256, 0, stream>>>(bufC, WprojT, proj_b, projF,
                                                    131072, 512, 512, EPI_F32);
  // residual + bug-compatible window_reverse -> x1 in d_out
  swin_resid<<<eb, TB, 0, stream>>>(x, projF, out);
  // BN2 + transpose to token-major bf16
  swin_bn_tok<<<eb, TB, 0, stream>>>(out, bn2_g, bn2_b, bn2_m, bn2_v, bufA);
  // MLP: GEMM1 + GELU, GEMM2
  swin_gemm_bf16<<<dim3(4, 1024), 256, 0, stream>>>(bufA, W1c, b1, h1,
                                                    131072, 512, 512, EPI_GELU);
  swin_gemm_bf16<<<dim3(4, 1024), 256, 0, stream>>>(h1, W2c, b2, bufC,
                                                    131072, 512, 512, EPI_BF16);
  // final residual: d_out = x1 + mlp_out
  swin_final_add<<<eb, TB, 0, stream>>>(bufC, out);
}
